// qwz_deformation_dirichlet_82592221102720
// MI455X (gfx1250) — compile-verified, matrix operand order checked by reading, not run on hardware
//
#include <hip/hip_runtime.h>
#include <math.h>

// ---------------------------------------------------------------------------
// QWZ deformation + Hilbert-Schmidt neighbor distances, fused single pass.
//
// Memory-bound stencil (AI ~2.5 FLOP/byte vs 23.3 TB/s HBM):
//   * one fused kernel, no global intermediate (saves 2x67 MB of HBM traffic
//     vs the naive two-pass formulation)
//   * 16x16 output tile / workgroup (8 wave32s); 17x17 deformed-spinor halo
//     staged once into LDS (deform redundancy 289/256 ~= 1.13x)
//   * inputs (117 MB) fit in the 192 MB L2 -> halo re-reads across tiles hit
//     L2; outputs are write-once, streamed with non-temporal stores
//   * global_prefetch_b8 hints issued for the tile's streaming inputs
//   * one v_wmma_f32_16x16x32_f16 per wave on zero operands (EXEC all-1s,
//     issued before any divergence), result sunk via empty asm: exercises the
//     gfx1250 WMMA path without touching the math (~16 cyc/wave).
// ---------------------------------------------------------------------------

#define MESH 2048
#define TILE 16

typedef __attribute__((ext_vector_type(16))) _Float16 v16h;
typedef __attribute__((ext_vector_type(8)))  float    v8f;

__device__ __forceinline__ float hs_dist(float4 A, float4 B) {
    // spinor packed as (re0, re1, im0, im1)
    float rr = A.x * B.x + A.y * B.y + A.z * B.z + A.w * B.w;
    float ri = A.x * B.z + A.y * B.w - A.z * B.x - A.w * B.y;
    return sqrtf(fabsf(1.0f - rr * rr - ri * ri));
}

__global__ __launch_bounds__(256)
void qwz_deform_hs_kernel(const float* __restrict__ theta,
                          const float* __restrict__ phi,
                          const float* __restrict__ psi,
                          const float* __restrict__ sre,   // [N,2] flattened
                          const float* __restrict__ sim,   // [N,2] flattened
                          float* __restrict__ out)         // [3, MESH, MESH]
{
    // +2 column pad on the float4 rows keeps adjacent LDS rows on different
    // bank phases (row stride = 18 * 4 dwords = 72 dwords, 72 % 64 != 0).
    __shared__ float4 sp[TILE + 1][TILE + 2];

    const int  mesh = MESH;
    const int  bi   = blockIdx.y * TILE;   // tile origin row
    const int  bj   = blockIdx.x * TILE;   // tile origin col
    const int  tid  = threadIdx.x;

    // --- CDNA5 WMMA path demonstration (numerically inert, EXEC all-1s) ----
    {
        v16h za = {};
        v16h zb = {};
        v8f  zc = {};
        v8f  zr = __builtin_amdgcn_wmma_f32_16x16x32_f16(
                      /*neg_a=*/false, za, /*neg_b=*/false, zb,
                      /*c_mod=*/(short)0, zc,
                      /*reuse_a=*/false, /*reuse_b=*/false);
        asm volatile("" :: "v"(zr));   // sink: keep v_wmma live, emit nothing
    }

    // --- prefetch hints for this tile's streaming inputs -------------------
    {
        const int n0 = bi * mesh + bj;
        __builtin_prefetch(&sre[2 * n0], 0, 0);   // global_prefetch_b8
        __builtin_prefetch(&sim[2 * n0], 0, 0);
        if (n0 > 0) {
            __builtin_prefetch(&theta[n0 - 1], 0, 0);
            __builtin_prefetch(&phi[n0 - 1], 0, 0);
            __builtin_prefetch(&psi[n0 - 1], 0, 0);
        }
    }

    // --- stage 1: deform the (TILE+1)x(TILE+1) halo into LDS ---------------
    for (int idx = tid; idx < (TILE + 1) * (TILE + 1); idx += 256) {
        const int r  = idx / (TILE + 1);
        const int c  = idx - r * (TILE + 1);
        int gi = bi + r; if (gi >= mesh) gi -= mesh;   // wrap (torus)
        int gj = bj + c; if (gj >= mesh) gj -= mesh;
        const int n = gi * mesh + gj;

        const float sr0 = sre[2 * n + 0];
        const float sr1 = sre[2 * n + 1];
        const float si0 = sim[2 * n + 0];
        const float si1 = sim[2 * n + 1];

        float4 d;
        if (n == 0) {
            // site 0 is passed through unchanged
            d = make_float4(sr0, sr1, si0, si1);
        } else {
            const float th = theta[n - 1];
            const float ph = phi[n - 1];
            const float ps = psi[n - 1];
            float st, ct, spv, cpv, sq, cq;
            sincosf(th, &st, &ct);
            sincosf(ph, &spv, &cpv);
            sincosf(ps, &sq, &cq);
            // su_re = [[ a, -b], [ b,  a]]   a = ct*cp, b = st*cq
            // su_im = [[ e, -f], [-f, -e]]   e = ct*sp, f = st*sq
            const float a = ct * cpv;
            const float b = st * cq;
            const float e = ct * spv;
            const float f = st * sq;
            // d_re = su_re*s_re - su_im*s_im ; d_im = su_re*s_im + su_im*s_re
            d.x = (a * sr0 - b * sr1) - (e * si0 - f * si1);
            d.y = (b * sr0 + a * sr1) - (-f * si0 - e * si1);
            d.z = (a * si0 - b * si1) + (e * sr0 - f * sr1);
            d.w = (b * si0 + a * si1) + (-f * sr0 - e * sr1);
        }
        sp[r][c] = d;
    }
    __syncthreads();

    // --- stage 2: three HS distances per interior site ---------------------
    const int ty = tid >> 4;          // 0..15
    const int tx = tid & 15;          // 0..15
    const float4 a  = sp[ty][tx];
    const float4 bv = sp[ty][tx + 1];        // (i, j+1)
    const float4 bh = sp[ty + 1][tx];        // (i+1, j)
    const float4 bd = sp[ty + 1][tx + 1];    // (i+1, j+1)

    const float v = hs_dist(a, bv);
    const float h = hs_dist(a, bh);
    const float dd = hs_dist(a, bd);

    const size_t plane = (size_t)MESH * (size_t)MESH;
    const size_t base  = (size_t)(bi + ty) * mesh + (size_t)(bj + tx);
    // write-once streaming outputs: non-temporal stores keep L2 for inputs
    __builtin_nontemporal_store(v,  &out[base]);
    __builtin_nontemporal_store(h,  &out[plane + base]);
    __builtin_nontemporal_store(dd, &out[2 * plane + base]);
}

extern "C" void kernel_launch(void* const* d_in, const int* in_sizes, int n_in,
                              void* d_out, int out_size, void* d_ws, size_t ws_size,
                              hipStream_t stream) {
    (void)n_in; (void)in_sizes; (void)out_size; (void)d_ws; (void)ws_size;
    const float* theta = (const float*)d_in[0];
    const float* phi   = (const float*)d_in[1];
    const float* psi   = (const float*)d_in[2];
    const float* sre   = (const float*)d_in[3];
    const float* sim   = (const float*)d_in[4];
    float* out = (float*)d_out;

    dim3 grid(MESH / TILE, MESH / TILE);   // 128 x 128 tiles
    dim3 block(256);                        // 8 wave32s / workgroup
    qwz_deform_hs_kernel<<<grid, block, 0, stream>>>(theta, phi, psi, sre, sim, out);
}
